// ManualGATLayer_62251255989135
// MI455X (gfx1250) — compile-verified, matrix operand order checked by reading, not run on hardware
//
#include <hip/hip_runtime.h>
#include <math.h>

typedef __attribute__((ext_vector_type(16))) _Float16 v16h;
typedef __attribute__((ext_vector_type(8)))  float    v8f;

#define N_NODES 50000
#define N_EDGES 1000000
#define IN_F    128
#define OUT_F   32
#define HEADS   4
#define NF      128   // HEADS*OUT_F == IN_F == 128
#define MTILE   80    // 50000 / 80 = 625 blocks exactly -> no bounds guards

// ---- workspace layout (byte offsets, all 16B-aligned) ----
#define WS_XTRANS 0ull          // 50000*128 f32  = 25,600,000 B
#define WS_SSRC   25600000ull   // 200000 f32     =    800,000 B
#define WS_SDST   26400000ull   // 200000 f32     =    800,000 B
#define WS_ESCORE 27200000ull   // 1M*4 f32       = 16,000,000 B
#define WS_HMAX   43200000ull   // 4 u32
#define WS_HSUM   43200032ull   // 4 f32
// total ~43.2 MB

// monotone order-preserving float<->uint encoding for atomicMax on floats
__device__ __forceinline__ unsigned fenc(float f) {
    unsigned u = __float_as_uint(f);
    return (u & 0x80000000u) ? ~u : (u | 0x80000000u);
}
__device__ __forceinline__ float fdec(unsigned k) {
    unsigned u = (k & 0x80000000u) ? (k & 0x7fffffffu) : ~k;
    return __uint_as_float(u);
}
__device__ __forceinline__ float lrelu(float v) { return v > 0.0f ? v : 0.2f * v; }

// ---------------- kernel 0: init out + reduction slots ----------------
__global__ void initK(float* __restrict__ out, int n,
                      unsigned* __restrict__ hmax, float* __restrict__ hsum) {
    int i = blockIdx.x * blockDim.x + threadIdx.x;
    for (; i < n; i += gridDim.x * blockDim.x) out[i] = 0.0f;
    if (blockIdx.x == 0 && threadIdx.x < HEADS) {
        hmax[threadIdx.x] = 0u;     // below any finite score's encoding
        hsum[threadIdx.x] = 0.0f;
    }
}

// ---------------- kernel 1: x_trans = x @ W.T via WMMA ----------------
// block = 256 threads = 8 waves; block computes MTILE(80) rows x 128 cols.
// wave w -> output cols [16w, 16w+16); per wave: 5 M-subtiles x 4 K-steps of
// v_wmma_f32_16x16x32_f16, with the 4 B fragments hoisted & reused across M.
__global__ __launch_bounds__(256) void gemmK(const float* __restrict__ x,
                                             const float* __restrict__ W,
                                             float* __restrict__ xt) {
    __shared__ _Float16 sA[MTILE * 132];   // x tile  [m][k], pitch 132
    __shared__ _Float16 sB[128 * 132];     // W       [n][k], pitch 132

    const int tid = threadIdx.x;
    const int m0  = blockIdx.x * MTILE;

    // stage W (128x128) as f16, [n][k]
#pragma unroll
    for (int i = 0; i < 64; ++i) {
        int e = tid + i * 256;
        int n = e >> 7, k = e & 127;
        sB[n * 132 + k] = (_Float16)W[n * 128 + k];
    }
    // stage x tile (80x128) as f16, [m][k]; grid divides N_NODES exactly
#pragma unroll
    for (int i = 0; i < (MTILE * 128) / 256; ++i) {
        int e = tid + i * 256;
        int r = e >> 7, k = e & 127;
        sA[r * 132 + k] = (_Float16)x[(size_t)(m0 + r) * IN_F + k];
    }
    __syncthreads();

    const int wv   = tid >> 5;
    const int lane = tid & 31;
    const int half = lane >> 4;     // K-group selector per ISA 16-bit A-layout
    const int mr   = lane & 15;     // A: row M; B: col N
    const int nb   = wv * 16;
    const int col  = nb + mr;

    // hoist B fragments: lanes 0-15 hold K 0-7 (V0-3) & 16-23 (V4-7);
    // lanes 16-31 hold K 8-15 & 24-31 (column-major per lane).
    const _Float16* pb = &sB[(nb + mr) * 132];
    v16h bf[4];
#pragma unroll
    for (int kb = 0; kb < 4; ++kb) {
#pragma unroll
        for (int e = 0; e < 16; ++e) {
            int k = kb * 32 + ((e >> 3) << 4) + half * 8 + (e & 7);
            bf[kb][e] = pb[k];
        }
    }

#pragma unroll
    for (int ms = 0; ms < MTILE / 16; ++ms) {
        const _Float16* pa = &sA[(ms * 16 + mr) * 132];
        v8f c = {0.f, 0.f, 0.f, 0.f, 0.f, 0.f, 0.f, 0.f};
#pragma unroll
        for (int kb = 0; kb < 4; ++kb) {
            v16h a;
#pragma unroll
            for (int e = 0; e < 16; ++e) {
                int k = kb * 32 + ((e >> 3) << 4) + half * 8 + (e & 7);
                a[e] = pa[k];
            }
            c = __builtin_amdgcn_wmma_f32_16x16x32_f16(
                    /*neg_a=*/false, a, /*neg_b=*/false, bf[kb],
                    /*c_mod=*/(short)0, c, /*reuse_a=*/false, /*reuse_b=*/false);
        }
        // C layout: VGPR j -> M = j + 8*half, N = lane&15 ; no bounds guard needed
        const size_t rbase = (size_t)(m0 + ms * 16 + half * 8) * NF + col;
#pragma unroll
        for (int j = 0; j < 8; ++j) {
            xt[rbase + (size_t)j * NF] = c[j];
        }
    }
}

// ---------------- kernel 2: per-node attention scalars ----------------
__global__ void nodeScoreK(const float* __restrict__ xt,
                           const float* __restrict__ a_src,
                           const float* __restrict__ a_dst,
                           float* __restrict__ ssrc, float* __restrict__ sdst) {
    int t = blockIdx.x * blockDim.x + threadIdx.x;
    if (t >= N_NODES * HEADS) return;
    int n = t >> 2, h = t & 3;
    const float4* row = (const float4*)(xt + (size_t)n * NF + h * OUT_F);
    const float4* av  = (const float4*)(a_src + h * OUT_F);
    const float4* bv  = (const float4*)(a_dst + h * OUT_F);
    float s = 0.0f, d = 0.0f;
#pragma unroll
    for (int i = 0; i < 8; ++i) {
        float4 xv = row[i], a = av[i], b = bv[i];
        s += xv.x * a.x + xv.y * a.y + xv.z * a.z + xv.w * a.w;
        d += xv.x * b.x + xv.y * b.y + xv.z * b.z + xv.w * b.w;
    }
    ssrc[t] = s;
    sdst[t] = d;
}

// ---------------- kernel 3a: edge scores + per-head global max ----------------
__global__ void edgeScoreMaxK(const int* __restrict__ eidx,
                              const float* __restrict__ ssrc,
                              const float* __restrict__ sdst,
                              float* __restrict__ escore,
                              unsigned* __restrict__ hmax) {
    const int stride = gridDim.x * blockDim.x;
    float4 m = make_float4(-3.0e38f, -3.0e38f, -3.0e38f, -3.0e38f);
    for (int e = blockIdx.x * blockDim.x + threadIdx.x; e < N_EDGES; e += stride) {
        int s = eidx[e], d = eidx[N_EDGES + e];
        float4 as = ((const float4*)ssrc)[s];
        float4 ad = ((const float4*)sdst)[d];
        float4 sc;
        sc.x = lrelu(as.x + ad.x);
        sc.y = lrelu(as.y + ad.y);
        sc.z = lrelu(as.z + ad.z);
        sc.w = lrelu(as.w + ad.w);
        ((float4*)escore)[e] = sc;
        m.x = fmaxf(m.x, sc.x); m.y = fmaxf(m.y, sc.y);
        m.z = fmaxf(m.z, sc.z); m.w = fmaxf(m.w, sc.w);
    }
    // wave32 shuffle reduction
    for (int off = 16; off > 0; off >>= 1) {
        m.x = fmaxf(m.x, __shfl_xor(m.x, off));
        m.y = fmaxf(m.y, __shfl_xor(m.y, off));
        m.z = fmaxf(m.z, __shfl_xor(m.z, off));
        m.w = fmaxf(m.w, __shfl_xor(m.w, off));
    }
    __shared__ float4 sm[8];
    if ((threadIdx.x & 31) == 0) sm[threadIdx.x >> 5] = m;
    __syncthreads();
    if (threadIdx.x == 0) {
        float4 r = sm[0];
        for (int i = 1; i < 8; ++i) {
            r.x = fmaxf(r.x, sm[i].x); r.y = fmaxf(r.y, sm[i].y);
            r.z = fmaxf(r.z, sm[i].z); r.w = fmaxf(r.w, sm[i].w);
        }
        atomicMax(&hmax[0], fenc(r.x));
        atomicMax(&hmax[1], fenc(r.y));
        atomicMax(&hmax[2], fenc(r.z));
        atomicMax(&hmax[3], fenc(r.w));
    }
}

// ---------------- kernel 3b: per-head exp-sum ----------------
__global__ void expSumK(const float* __restrict__ escore,
                        const unsigned* __restrict__ hmax,
                        float* __restrict__ hsum) {
    float4 mx;
    mx.x = fdec(hmax[0]); mx.y = fdec(hmax[1]);
    mx.z = fdec(hmax[2]); mx.w = fdec(hmax[3]);
    const int stride = gridDim.x * blockDim.x;
    float4 acc = make_float4(0.f, 0.f, 0.f, 0.f);
    for (int e = blockIdx.x * blockDim.x + threadIdx.x; e < N_EDGES; e += stride) {
        float4 sc = ((const float4*)escore)[e];
        acc.x += __expf(sc.x - mx.x);
        acc.y += __expf(sc.y - mx.y);
        acc.z += __expf(sc.z - mx.z);
        acc.w += __expf(sc.w - mx.w);
    }
    for (int off = 16; off > 0; off >>= 1) {
        acc.x += __shfl_xor(acc.x, off);
        acc.y += __shfl_xor(acc.y, off);
        acc.z += __shfl_xor(acc.z, off);
        acc.w += __shfl_xor(acc.w, off);
    }
    __shared__ float4 sm[8];
    if ((threadIdx.x & 31) == 0) sm[threadIdx.x >> 5] = acc;
    __syncthreads();
    if (threadIdx.x == 0) {
        float4 r = sm[0];
        for (int i = 1; i < 8; ++i) {
            r.x += sm[i].x; r.y += sm[i].y; r.z += sm[i].z; r.w += sm[i].w;
        }
        atomicAdd(&hsum[0], r.x);
        atomicAdd(&hsum[1], r.y);
        atomicAdd(&hsum[2], r.z);
        atomicAdd(&hsum[3], r.w);
    }
}

// ---------------- kernel 4: weighted gather + atomic scatter-add ----------------
// one wave per edge (grid divides E exactly); lane l -> cols [4l,4l+4), head = l>>3.
__global__ __launch_bounds__(256) void scatterK(const int* __restrict__ eidx,
                                                const float* __restrict__ xt,
                                                const float* __restrict__ escore,
                                                const unsigned* __restrict__ hmax,
                                                const float* __restrict__ hsum,
                                                float* __restrict__ out) {
    const int wave = (blockIdx.x * blockDim.x + threadIdx.x) >> 5;
    const int lane = threadIdx.x & 31;
    const int s = eidx[wave];
    const int d = eidx[N_EDGES + wave];
    const int h = lane >> 3;
    const float w = __expf(escore[wave * HEADS + h] - fdec(hmax[h])) / hsum[h];
    const float4 v = ((const float4*)(xt + (size_t)s * NF))[lane];   // b128 gather (L2-resident)
    float* o = out + (size_t)d * NF + lane * 4;
    atomicAdd(o + 0, w * v.x);
    atomicAdd(o + 1, w * v.y);
    atomicAdd(o + 2, w * v.z);
    atomicAdd(o + 3, w * v.w);
}

extern "C" void kernel_launch(void* const* d_in, const int* in_sizes, int n_in,
                              void* d_out, int out_size, void* d_ws, size_t ws_size,
                              hipStream_t stream) {
    const float* x     = (const float*)d_in[0];
    const int*   eidx  = (const int*)d_in[1];     // [2, E] row-major
    const float* W     = (const float*)d_in[2];
    const float* a_src = (const float*)d_in[3];   // flat [HEADS*OUT_F]
    const float* a_dst = (const float*)d_in[4];
    float* out = (float*)d_out;

    char* ws = (char*)d_ws;
    float*    xt     = (float*)(ws + WS_XTRANS);
    float*    ssrc   = (float*)(ws + WS_SSRC);
    float*    sdst   = (float*)(ws + WS_SDST);
    float*    escore = (float*)(ws + WS_ESCORE);
    unsigned* hmax   = (unsigned*)(ws + WS_HMAX);
    float*    hsum   = (float*)(ws + WS_HSUM);

    initK<<<256, 256, 0, stream>>>(out, N_NODES * NF, hmax, hsum);
    gemmK<<<N_NODES / MTILE, 256, 0, stream>>>(x, W, xt);
    nodeScoreK<<<(N_NODES * HEADS + 255) / 256, 256, 0, stream>>>(xt, a_src, a_dst, ssrc, sdst);
    edgeScoreMaxK<<<512, 256, 0, stream>>>(eidx, ssrc, sdst, escore, hmax);
    expSumK<<<512, 256, 0, stream>>>(escore, hmax, hsum);
    scatterK<<<(N_EDGES * 32) / 256, 256, 0, stream>>>(eidx, xt, escore, hmax, hsum, out);
}
